// GATNet_89481348645141
// MI455X (gfx1250) — compile-verified, matrix operand order checked by reading, not run on hardware
//
#include <hip/hip_runtime.h>
#include <hip/hip_bf16.h>
#include <math.h>

typedef __attribute__((ext_vector_type(16))) _Float16 v16h;
typedef __attribute__((ext_vector_type(8)))  float    v8f;

#define F1  128   // input features
#define H1h 8     // heads layer1
#define D1d 8     // dim per head layer1
#define FH  64    // H1*D1
#define C2  16    // classes
#define NEG_SLOPE 0.2f

__device__ __forceinline__ void atomicMaxF(float* addr, float val) {
  // signed-int compare works for non-negative floats, unsigned-min for negative
  if (val >= 0.0f) atomicMax((int*)addr, __float_as_int(val));
  else             atomicMin((unsigned int*)addr, __float_as_uint(val));
}

__device__ __forceinline__ void edge_sd(const int* __restrict__ ei, int eidx, int E,
                                        int& s, int& d) {
  if (eidx < E) { s = ei[eidx]; d = ei[E + eidx]; }
  else          { s = eidx - E; d = s; }            // self-loop
}

__global__ void fill_kernel(float* __restrict__ p, float v, int n) {
  int i = blockIdx.x * blockDim.x + threadIdx.x;
  if (i < n) p[i] = v;
}

// h[N,64] = x[N,128] @ W[128,64], one 16x16 tile per wave, f16 WMMA / f32 acc
__global__ void gemm1_wmma(const float* __restrict__ x, const float* __restrict__ W,
                           float* __restrict__ h, int N) {
  int wid  = (blockIdx.x * blockDim.x + threadIdx.x) >> 5;
  int lane = threadIdx.x & 31;
  int nTileRows = (N + 15) >> 4;
  if (wid >= nTileRows * 4) return;               // wave-uniform exit
  int tr = wid >> 2, tc = wid & 3;
  int m  = lane & 15, hi = lane >> 4;
  int row = tr * 16 + m; if (row >= N) row = N - 1;
  int col = tc * 16 + m;
  v8f acc = {};
  for (int k0 = 0; k0 < F1; k0 += 32) {
    v16h a, b;
    const float* xr = x + (long long)row * F1 + k0 + hi * 8;
    const float* wc = W + (k0 + hi * 8) * FH + col;
#pragma unroll
    for (int i = 0; i < 8; ++i) {
      a[i]     = (_Float16)xr[i];          // K = k0 + hi*8 + i
      a[i + 8] = (_Float16)xr[16 + i];     // K = k0 + 16 + hi*8 + i
      b[i]     = (_Float16)wc[i * FH];
      b[i + 8] = (_Float16)wc[(16 + i) * FH];
    }
    acc = __builtin_amdgcn_wmma_f32_16x16x32_f16(false, a, false, b,
                                                 (short)0, acc, false, false);
  }
#pragma unroll
  for (int r = 0; r < 8; ++r) {
    int orow = tr * 16 + r + hi * 8;
    if (orow < N) h[(long long)orow * FH + col] = acc[r];
  }
}

// h[N,16] = x1[N,64] @ W[64,16]
__global__ void gemm2_wmma(const float* __restrict__ x1, const float* __restrict__ W,
                           float* __restrict__ h, int N) {
  int wid  = (blockIdx.x * blockDim.x + threadIdx.x) >> 5;
  int lane = threadIdx.x & 31;
  int nTileRows = (N + 15) >> 4;
  if (wid >= nTileRows) return;
  int m = lane & 15, hi = lane >> 4;
  int row = wid * 16 + m; if (row >= N) row = N - 1;
  int col = m;
  v8f acc = {};
  for (int k0 = 0; k0 < FH; k0 += 32) {
    v16h a, b;
    const float* xr = x1 + (long long)row * FH + k0 + hi * 8;
    const float* wc = W + (k0 + hi * 8) * C2 + col;
#pragma unroll
    for (int i = 0; i < 8; ++i) {
      a[i]     = (_Float16)xr[i];
      a[i + 8] = (_Float16)xr[16 + i];
      b[i]     = (_Float16)wc[i * C2];
      b[i + 8] = (_Float16)wc[(16 + i) * C2];
    }
    acc = __builtin_amdgcn_wmma_f32_16x16x32_f16(false, a, false, b,
                                                 (short)0, acc, false, false);
  }
#pragma unroll
  for (int r = 0; r < 8; ++r) {
    int orow = wid * 16 + r + hi * 8;
    if (orow < N) h[(long long)orow * C2 + col] = acc[r];
  }
}

// per (node, head): alpha_src/alpha_dst dot products
__global__ void alpha1_kernel(const float* __restrict__ h1,
                              const float* __restrict__ att_s,
                              const float* __restrict__ att_d,
                              float* __restrict__ as1, float* __restrict__ ad1, int N) {
  int t = blockIdx.x * blockDim.x + threadIdx.x;
  if (t >= N * H1h) return;
  int n = t >> 3, hh = t & 7;
  const float* hp = h1 + (long long)n * FH + hh * D1d;
  float s = 0.f, d = 0.f;
#pragma unroll
  for (int i = 0; i < D1d; ++i) {
    s += hp[i] * att_s[hh * D1d + i];
    d += hp[i] * att_d[hh * D1d + i];
  }
  as1[t] = s; ad1[t] = d;
}

__global__ void e1_max(const int* __restrict__ ei, const float* __restrict__ as1,
                       const float* __restrict__ ad1, float* __restrict__ m1,
                       int E, int N) {
  int t = blockIdx.x * blockDim.x + threadIdx.x;
  if (t >= (E + N) * H1h) return;
  int eidx = t >> 3, hh = t & 7, s, d;
  edge_sd(ei, eidx, E, s, d);
  float e = as1[s * H1h + hh] + ad1[d * H1h + hh];
  e = (e >= 0.f) ? e : NEG_SLOPE * e;
  atomicMaxF(&m1[d * H1h + hh], e);
}

__global__ void e1_sum(const int* __restrict__ ei, const float* __restrict__ as1,
                       const float* __restrict__ ad1, const float* __restrict__ m1,
                       float* __restrict__ dn1, int E, int N) {
  int t = blockIdx.x * blockDim.x + threadIdx.x;
  if (t >= (E + N) * H1h) return;
  int eidx = t >> 3, hh = t & 7, s, d;
  edge_sd(ei, eidx, E, s, d);
  float e = as1[s * H1h + hh] + ad1[d * H1h + hh];
  e = (e >= 0.f) ? e : NEG_SLOPE * e;
  atomicAdd(&dn1[d * H1h + hh], __expf(e - m1[d * H1h + hh]));
}

__global__ void e1_agg(const int* __restrict__ ei, const float* __restrict__ as1,
                       const float* __restrict__ ad1, const float* __restrict__ m1,
                       const float* __restrict__ dn1, const float* __restrict__ h1,
                       float* __restrict__ agg1, int E, int N) {
  int t = blockIdx.x * blockDim.x + threadIdx.x;
  if (t >= (E + N) * H1h) return;
  int eidx = t >> 3, hh = t & 7, s, d;
  edge_sd(ei, eidx, E, s, d);
  float e = as1[s * H1h + hh] + ad1[d * H1h + hh];
  e = (e >= 0.f) ? e : NEG_SLOPE * e;
  float alpha = __expf(e - m1[d * H1h + hh]) / (dn1[d * H1h + hh] + 1e-16f);
  const float4* hv = (const float4*)(h1 + (long long)s * FH + hh * D1d);
  float4 a0 = hv[0], a1 = hv[1];
  float* o = agg1 + (long long)d * FH + hh * D1d;
  atomicAdd(o + 0, a0.x * alpha); atomicAdd(o + 1, a0.y * alpha);
  atomicAdd(o + 2, a0.z * alpha); atomicAdd(o + 3, a0.w * alpha);
  atomicAdd(o + 4, a1.x * alpha); atomicAdd(o + 5, a1.y * alpha);
  atomicAdd(o + 6, a1.z * alpha); atomicAdd(o + 7, a1.w * alpha);
}

__global__ void bias_elu(float* __restrict__ a, const float* __restrict__ b, int N) {
  int t = blockIdx.x * blockDim.x + threadIdx.x;
  if (t >= N * FH) return;
  float v = a[t] + b[t & (FH - 1)];
  a[t] = (v > 0.f) ? v : (__expf(v) - 1.0f);
}

__global__ void alpha2_kernel(const float* __restrict__ h2,
                              const float* __restrict__ att_s,
                              const float* __restrict__ att_d,
                              float* __restrict__ as2, float* __restrict__ ad2, int N) {
  int n = blockIdx.x * blockDim.x + threadIdx.x;
  if (n >= N) return;
  const float* hp = h2 + (long long)n * C2;
  float s = 0.f, d = 0.f;
#pragma unroll
  for (int c = 0; c < C2; ++c) { s += hp[c] * att_s[c]; d += hp[c] * att_d[c]; }
  as2[n] = s; ad2[n] = d;
}

__global__ void e2_max(const int* __restrict__ ei, const float* __restrict__ as2,
                       const float* __restrict__ ad2, float* __restrict__ m2,
                       int E, int N) {
  int t = blockIdx.x * blockDim.x + threadIdx.x;
  if (t >= E + N) return;
  int s, d; edge_sd(ei, t, E, s, d);
  float e = as2[s] + ad2[d];
  e = (e >= 0.f) ? e : NEG_SLOPE * e;
  atomicMaxF(&m2[d], e);
}

__global__ void e2_sum(const int* __restrict__ ei, const float* __restrict__ as2,
                       const float* __restrict__ ad2, const float* __restrict__ m2,
                       float* __restrict__ dn2, int E, int N) {
  int t = blockIdx.x * blockDim.x + threadIdx.x;
  if (t >= E + N) return;
  int s, d; edge_sd(ei, t, E, s, d);
  float e = as2[s] + ad2[d];
  e = (e >= 0.f) ? e : NEG_SLOPE * e;
  atomicAdd(&dn2[d], __expf(e - m2[d]));
}

__global__ void e2_agg(const int* __restrict__ ei, const float* __restrict__ as2,
                       const float* __restrict__ ad2, const float* __restrict__ m2,
                       const float* __restrict__ dn2, const float* __restrict__ h2,
                       float* __restrict__ agg2, int E, int N) {
  int t = blockIdx.x * blockDim.x + threadIdx.x;
  if (t >= E + N) return;
  int s, d; edge_sd(ei, t, E, s, d);
  float e = as2[s] + ad2[d];
  e = (e >= 0.f) ? e : NEG_SLOPE * e;
  float alpha = __expf(e - m2[d]) / (dn2[d] + 1e-16f);
  const float4* hv = (const float4*)(h2 + (long long)s * C2);
  float* o = agg2 + (long long)d * C2;
#pragma unroll
  for (int q = 0; q < 4; ++q) {
    float4 a = hv[q];
    atomicAdd(o + 4 * q + 0, a.x * alpha); atomicAdd(o + 4 * q + 1, a.y * alpha);
    atomicAdd(o + 4 * q + 2, a.z * alpha); atomicAdd(o + 4 * q + 3, a.w * alpha);
  }
}

__global__ void final_logsoftmax(const float* __restrict__ agg2,
                                 const float* __restrict__ b2,
                                 float* __restrict__ out, int N) {
  int n = blockIdx.x * blockDim.x + threadIdx.x;
  if (n >= N) return;
  float v[C2]; float mx = -INFINITY;
#pragma unroll
  for (int c = 0; c < C2; ++c) { v[c] = agg2[(long long)n * C2 + c] + b2[c]; mx = fmaxf(mx, v[c]); }
  float sum = 0.f;
#pragma unroll
  for (int c = 0; c < C2; ++c) sum += __expf(v[c] - mx);
  float lse = mx + __logf(sum);
#pragma unroll
  for (int c = 0; c < C2; ++c) out[(long long)n * C2 + c] = v[c] - lse;
}

extern "C" void kernel_launch(void* const* d_in, const int* in_sizes, int n_in,
                              void* d_out, int out_size, void* d_ws, size_t ws_size,
                              hipStream_t stream) {
  const float* x   = (const float*)d_in[0];
  const int*   ei  = (const int*)  d_in[1];
  const float* W1  = (const float*)d_in[2];
  const float* aS1 = (const float*)d_in[3];
  const float* aD1 = (const float*)d_in[4];
  const float* b1  = (const float*)d_in[5];
  const float* W2  = (const float*)d_in[6];
  const float* aS2 = (const float*)d_in[7];
  const float* aD2 = (const float*)d_in[8];
  const float* b2  = (const float*)d_in[9];
  float* out = (float*)d_out;

  const int N = in_sizes[0] / F1;
  const int E = in_sizes[1] / 2;
  const long long Np = N;

  float* ws   = (float*)d_ws;
  float* h1   = ws;               // N*64
  float* agg1 = h1   + Np * 64;   // N*64  (becomes x1 after bias+ELU)
  float* dn1  = agg1 + Np * 64;   // N*8
  float* agg2 = dn1  + Np * 8;    // N*16
  float* dn2  = agg2 + Np * 16;   // N
  float* m1   = dn2  + Np;        // N*8
  float* m2   = m1   + Np * 8;    // N
  float* as1  = m2   + Np;        // N*8
  float* ad1  = as1  + Np * 8;    // N*8
  float* h2   = ad1  + Np * 8;    // N*16
  float* as2  = h2   + Np * 16;   // N
  float* ad2  = as2  + Np;        // N

  const int B = 256;
  auto cdiv = [](long long a, int b) { return (int)((a + b - 1) / b); };

  // init: zeros for [agg1, dn1, agg2, dn2]; -inf for [m1, m2] (contiguous runs)
  long long zc = Np * (64 + 8 + 16 + 1);
  fill_kernel<<<cdiv(zc, B), B, 0, stream>>>(agg1, 0.0f, (int)zc);
  long long ic = Np * (8 + 1);
  fill_kernel<<<cdiv(ic, B), B, 0, stream>>>(m1, -INFINITY, (int)ic);

  // ---- layer 1 ----
  long long waves1 = ((N + 15) / 16) * 4LL;
  gemm1_wmma<<<cdiv(waves1 * 32, B), B, 0, stream>>>(x, W1, h1, N);
  alpha1_kernel<<<cdiv(Np * H1h, B), B, 0, stream>>>(h1, aS1, aD1, as1, ad1, N);
  long long T1 = (long long)(E + N) * H1h;
  e1_max<<<cdiv(T1, B), B, 0, stream>>>(ei, as1, ad1, m1, E, N);
  e1_sum<<<cdiv(T1, B), B, 0, stream>>>(ei, as1, ad1, m1, dn1, E, N);
  e1_agg<<<cdiv(T1, B), B, 0, stream>>>(ei, as1, ad1, m1, dn1, h1, agg1, E, N);
  bias_elu<<<cdiv(Np * FH, B), B, 0, stream>>>(agg1, b1, N);

  // ---- layer 2 ----
  long long waves2 = (N + 15) / 16;
  gemm2_wmma<<<cdiv(waves2 * 32, B), B, 0, stream>>>(agg1, W2, h2, N);
  alpha2_kernel<<<cdiv(Np, B), B, 0, stream>>>(h2, aS2, aD2, as2, ad2, N);
  long long T2 = E + N;
  e2_max<<<cdiv(T2, B), B, 0, stream>>>(ei, as2, ad2, m2, E, N);
  e2_sum<<<cdiv(T2, B), B, 0, stream>>>(ei, as2, ad2, m2, dn2, E, N);
  e2_agg<<<cdiv(T2, B), B, 0, stream>>>(ei, as2, ad2, m2, dn2, h2, agg2, E, N);

  final_logsoftmax<<<cdiv(Np, B), B, 0, stream>>>(agg2, b2, out, N);
}
